// LookAheadTransformer_53257594470917
// MI455X (gfx1250) — matrix operand order repaired and checked by reading, not compiled, not run on hardware
//
#include <hip/hip_runtime.h>
#include <hip/hip_bf16.h>
#include <math.h>

// Problem constants (fixed by the reference)
#define BB_   2
#define TT_   2048
#define DD_   1024
#define HH_   16
#define HD_   64
#define D3_   3072
#define DFF_  4096
#define NR_   4096          // B*T token rows
#define WIN_  10

typedef __bf16 bf16_t;
typedef __attribute__((ext_vector_type(16))) __bf16 bf16x16;
typedef __attribute__((ext_vector_type(8)))  __bf16 bf16x8;
typedef __attribute__((ext_vector_type(8)))  float  f32x8;

// ---------------------------------------------------------------------------
// WMMA: D = A(16x32 bf16) * B(32x16 bf16) + C(16x16 f32), wave32
// ---------------------------------------------------------------------------
__device__ __forceinline__ f32x8 wmma_bf16f32(bf16x16 a, bf16x16 b, f32x8 c) {
  return __builtin_amdgcn_wmma_f32_16x16x32_bf16(false, a, false, b,
                                                 (short)0, c, false, false);
}

// A-fragment (16x32, 16-bit): lanes 0-15 hold row M=lane, K={0..7,16..23};
// lanes 16-31 hold row M=lane-16, K={8..15,24..31}. base -> A[row0][k0], ld elems.
__device__ __forceinline__ bf16x16 load_a_frag(const bf16_t* base, int ld, int lane) {
  const int m  = lane & 15;
  const int kh = (lane >> 4) ? 8 : 0;
  const bf16_t* p = base + (size_t)m * ld + kh;
  union { bf16x16 f; bf16x8 h[2]; } u;
  u.h[0] = *(const bf16x8*)(p);        // K = kh .. kh+7
  u.h[1] = *(const bf16x8*)(p + 16);   // K = kh+16 .. kh+23
  return u.f;
}

// B-fragment (32x16, 16-bit): lane n=lane&15 holds column n; lanes 0-15 K={0..7,16..23},
// lanes 16-31 K={8..15,24..31}, consecutive K packed 2/VGPR.  B[k][n] = W[n][k] for a
// row-major W[N,K], so each lane reads two 16-byte chunks. base -> W[col0][k0].
__device__ __forceinline__ bf16x16 load_b_frag(const bf16_t* base, int ld, int lane) {
  const int n  = lane & 15;
  const int kh = (lane >> 4) ? 8 : 0;
  const bf16_t* p = base + (size_t)n * ld + kh;
  union { bf16x16 f; bf16x8 h[2]; } u;
  u.h[0] = *(const bf16x8*)(p);
  u.h[1] = *(const bf16x8*)(p + 16);
  return u.f;
}

// ---------------------------------------------------------------------------
// f32 -> bf16 cast (weights; done once per launch, then resident in 192MB L2)
// ---------------------------------------------------------------------------
__global__ void cvt_f32_bf16(const float* __restrict__ in, bf16_t* __restrict__ out, int n) {
  int i = blockIdx.x * blockDim.x + threadIdx.x;
  if (i < n) out[i] = (bf16_t)in[i];
}

// ---------------------------------------------------------------------------
// LayerNorm(x) * g + b -> bf16, one row (D=1024) per 256-thread block
// ---------------------------------------------------------------------------
__global__ __launch_bounds__(256)
void ln_bf16(const float* __restrict__ x, const float* __restrict__ g,
             const float* __restrict__ b, bf16_t* __restrict__ out) {
  __shared__ float red[16];
  const int row = blockIdx.x;
  const float* xr = x + (size_t)row * DD_;
  float v[4], s = 0.f, ss = 0.f;
#pragma unroll
  for (int i = 0; i < 4; ++i) {
    v[i] = xr[threadIdx.x + 256 * i];
    s += v[i]; ss += v[i] * v[i];
  }
#pragma unroll
  for (int m = 16; m > 0; m >>= 1) {
    s  += __shfl_xor(s,  m, 32);
    ss += __shfl_xor(ss, m, 32);
  }
  const int wave = threadIdx.x >> 5;
  if ((threadIdx.x & 31) == 0) { red[wave] = s; red[8 + wave] = ss; }
  __syncthreads();
  float S = 0.f, SS = 0.f;
#pragma unroll
  for (int w = 0; w < 8; ++w) { S += red[w]; SS += red[8 + w]; }
  const float mean = S * (1.f / DD_);
  const float var  = SS * (1.f / DD_) - mean * mean;
  const float inv  = rsqrtf(var + 1e-5f);
#pragma unroll
  for (int i = 0; i < 4; ++i) {
    const int c = threadIdx.x + 256 * i;
    out[(size_t)row * DD_ + c] = (bf16_t)((v[i] - mean) * inv * g[c] + b[c]);
  }
}

// ---------------------------------------------------------------------------
// Per-head V transpose: qkv v-slice [token][hd] -> vT [b,h][hd][token] (bf16)
// One block per (b, h, 64-token chunk); classic LDS tile transpose.
// ---------------------------------------------------------------------------
__global__ __launch_bounds__(256)
void transpose_v(const bf16_t* __restrict__ qkv, bf16_t* __restrict__ vT) {
  __shared__ bf16_t tile[64][68];           // +4 pad vs 64-bank LDS
  const int bid    = blockIdx.x;            // B*H*(T/64) = 1024
  const int tchunk = bid & 31;
  const int hh     = (bid >> 5) & 15;
  const int bb     = bid >> 9;
  const int t0     = tchunk * 64;

  const bf16_t* src = qkv + (size_t)(bb * TT_ + t0) * D3_ + 2 * DD_ + hh * HD_;
#pragma unroll
  for (int i = 0; i < 16; ++i) {
    const int e = threadIdx.x + 256 * i;    // 0..4095
    const int r = e >> 6, c = e & 63;       // r = token, c = hd
    tile[r][c] = src[(size_t)r * D3_ + c];
  }
  __syncthreads();
  bf16_t* dst = vT + (size_t)(bb * HH_ + hh) * HD_ * TT_ + t0;
#pragma unroll
  for (int i = 0; i < 16; ++i) {
    const int e = threadIdx.x + 256 * i;
    const int r = e >> 6, c = e & 63;       // r = hd, c = token
    dst[(size_t)r * TT_ + c] = tile[c][r];
  }
}

// ---------------------------------------------------------------------------
// Generic WMMA GEMM: C[M,N] = A[M,K](bf16) * W[N,K]^T (bf16) + bias
// wave tile 32x64 (2x4 WMMA tiles), block = 8 waves -> 128x128 tile.
// EPI 0: store bf16    EPI 1: f32 = resid + C    EPI 2: exact-erf GELU -> bf16
// ---------------------------------------------------------------------------
template <int EPI>
__global__ __launch_bounds__(256)
void gemm_wmma(const bf16_t* __restrict__ A, const bf16_t* __restrict__ Wt,
               const float* __restrict__ bias, const float* __restrict__ resid,
               float* __restrict__ outF, bf16_t* __restrict__ outB,
               int M, int N, int K) {
  const int lane = threadIdx.x & 31;
  const int wave = threadIdx.x >> 5;
  const int wr   = wave & 3;      // 4 row-waves
  const int wc   = wave >> 2;     // 2 col-waves
  const int row0 = blockIdx.y * 128 + wr * 32;
  const int col0 = blockIdx.x * 128 + wc * 64;

  const bf16_t* Abase = A  + (size_t)row0 * K;
  const bf16_t* Wbase = Wt + (size_t)col0 * K;

  f32x8 c[2][4] = {};
  for (int k0 = 0; k0 < K; k0 += 32) {
    if (k0 + 32 < K) {   // warm the next K-step through the cache path
      __builtin_prefetch(Abase + (size_t)(lane & 15) * K + k0 + 32, 0, 1);
      __builtin_prefetch(Wbase + (size_t)(lane & 15) * K + k0 + 32, 0, 1);
    }
    bf16x16 a0 = load_a_frag(Abase + k0, K, lane);
    bf16x16 a1 = load_a_frag(Abase + (size_t)16 * K + k0, K, lane);
    bf16x16 bf[4];
#pragma unroll
    for (int j = 0; j < 4; ++j)
      bf[j] = load_b_frag(Wbase + (size_t)(16 * j) * K + k0, K, lane);
#pragma unroll
    for (int j = 0; j < 4; ++j) {
      c[0][j] = wmma_bf16f32(a0, bf[j], c[0][j]);
      c[1][j] = wmma_bf16f32(a1, bf[j], c[1][j]);
    }
  }

  const int half = lane >> 4;
#pragma unroll
  for (int i = 0; i < 2; ++i)
#pragma unroll
    for (int j = 0; j < 4; ++j)
#pragma unroll
      for (int r = 0; r < 8; ++r) {
        const int row = row0 + i * 16 + r + 8 * half;     // C layout: M=r+8*(lane>=16)
        const int col = col0 + j * 16 + (lane & 15);      //           N=lane&15
        float v = c[i][j][r] + bias[col];
        const size_t idx = (size_t)row * N + col;
        if constexpr (EPI == 0) {
          outB[idx] = (bf16_t)v;
        } else if constexpr (EPI == 1) {
          outF[idx] = resid[idx] + v;
        } else {
          v = 0.5f * v * (1.f + erff(v * 0.70710678118654752f));  // exact GELU
          outB[idx] = (bf16_t)v;
        }
      }
}

// ---------------------------------------------------------------------------
// Flash-style look-ahead attention. One wave per (b, h, 32-query tile).
// qkv: bf16 [NR_, 3072] rows = (b*T+t), cols = [q | k | v] each H*HD.
// vT:  bf16 [B*H][HD][T]  (pre-transposed V, contiguous B-fragment loads)
// o:   bf16 [NR_, 1024]
// Per 32-key block: 8 WMMA (scores) + 8 WMMA (P.V) against 8 fragment loads;
// K/V fragments are shared across both 16-row tiles.
// ---------------------------------------------------------------------------
__global__ __launch_bounds__(256)
void attn_wmma(const bf16_t* __restrict__ qkv, const bf16_t* __restrict__ vT,
               bf16_t* __restrict__ o) {
  __shared__ bf16_t pbuf[8][32 * 32];       // per-wave P relay (A-layout), 2KB

  const int lane = threadIdx.x & 31;
  const int wave = threadIdx.x >> 5;
  const int task = blockIdx.x * 8 + wave;   // 2048 tasks = B*H*(T/32)
  const int qt = task & 63;
  const int hh = (task >> 6) & 15;
  const int bb = task >> 10;
  const int q0 = qt * 32;
  const int half = lane >> 4;

  const bf16_t* qbase = qkv + (size_t)bb * TT_ * D3_ + hh * HD_;
  const bf16_t* kbase = qbase + DD_;
  const bf16_t* vtb   = vT + (size_t)(bb * HH_ + hh) * HD_ * TT_;

  // Q fragments: 2 row-tiles x 2 hd-chunks (K=32 each)
  bf16x16 qa[2][2];
#pragma unroll
  for (int i = 0; i < 2; ++i) {
    qa[i][0] = load_a_frag(qbase + (size_t)(q0 + 16 * i) * D3_,      D3_, lane);
    qa[i][1] = load_a_frag(qbase + (size_t)(q0 + 16 * i) * D3_ + 32, D3_, lane);
  }

  f32x8 oacc[2][4] = {};
  float mrow[2][8], lrow[2][8];
#pragma unroll
  for (int i = 0; i < 2; ++i)
#pragma unroll
    for (int r = 0; r < 8; ++r) { mrow[i][r] = -3.0e38f; lrow[i][r] = 0.f; }

  int kmax = q0 + 31 + WIN_;
  if (kmax > TT_ - 1) kmax = TT_ - 1;
  const int nkb = (kmax >> 5) + 1;          // 32-key blocks (band mask -> ~i/32)

  for (int kb = 0; kb < nkb; ++kb) {
    const int key0 = kb * 32;

    // Scores for two 16-key sub-tiles; K-fragments shared across row-tiles
#pragma unroll
    for (int sub = 0; sub < 2; ++sub) {
      const int key16 = key0 + sub * 16;
      const bf16x16 kf0 = load_b_frag(kbase + (size_t)key16 * D3_,      D3_, lane);
      const bf16x16 kf1 = load_b_frag(kbase + (size_t)key16 * D3_ + 32, D3_, lane);
      const int kj = key16 + (lane & 15);
#pragma unroll
      for (int i = 0; i < 2; ++i) {
        f32x8 s = {};
        s = wmma_bf16f32(qa[i][0], kf0, s);
        s = wmma_bf16f32(qa[i][1], kf1, s);
#pragma unroll
        for (int r = 0; r < 8; ++r) {
          const int qi = q0 + 16 * i + r + 8 * half;
          float sv = s[r] * 0.125f;                     // 1/sqrt(64)
          sv = (kj <= qi + WIN_) ? sv : -3.0e38f;       // look-ahead band mask

          float mx = sv;                                // row max over 16-lane group
#pragma unroll
          for (int d = 8; d > 0; d >>= 1) mx = fmaxf(mx, __shfl_xor(mx, d, 16));
          const float mnew  = fmaxf(mrow[i][r], mx);
          const float alpha = __expf(mrow[i][r] - mnew);
          const float p     = __expf(sv - mnew);
          float psum = p;
#pragma unroll
          for (int d = 8; d > 0; d >>= 1) psum += __shfl_xor(psum, d, 16);

          lrow[i][r] = lrow[i][r] * alpha + psum;
          mrow[i][r] = mnew;
#pragma unroll
          for (int t = 0; t < 4; ++t) oacc[i][t][r] *= alpha;

          pbuf[wave][(16 * i + r + 8 * half) * 32 + sub * 16 + (lane & 15)] = (bf16_t)p;
        }
      }
    }

    // P.V: P tiles as A-fragments from LDS, V columns contiguous from vT
    const bf16x16 pa0 = load_a_frag(&pbuf[wave][0],       32, lane);
    const bf16x16 pa1 = load_a_frag(&pbuf[wave][16 * 32], 32, lane);
#pragma unroll
    for (int t = 0; t < 4; ++t) {
      const bf16x16 vb = load_b_frag(vtb + (size_t)(t * 16) * TT_ + key0, TT_, lane);
      oacc[0][t] = wmma_bf16f32(pa0, vb, oacc[0][t]);
      oacc[1][t] = wmma_bf16f32(pa1, vb, oacc[1][t]);
    }
  }

  // normalize and scatter heads back to [NR_, D]
#pragma unroll
  for (int i = 0; i < 2; ++i)
#pragma unroll
    for (int t = 0; t < 4; ++t)
#pragma unroll
      for (int r = 0; r < 8; ++r) {
        const int row = q0 + 16 * i + r + 8 * half;
        const int col = hh * HD_ + t * 16 + (lane & 15);
        o[(size_t)(bb * TT_ + row) * DD_ + col] = (bf16_t)(oacc[i][t][r] / lrow[i][r]);
      }
}

// ---------------------------------------------------------------------------
// Launch
// ---------------------------------------------------------------------------
extern "C" void kernel_launch(void* const* d_in, const int* in_sizes, int n_in,
                              void* d_out, int out_size, void* d_ws, size_t ws_size,
                              hipStream_t stream) {
  const float* x        = (const float*)d_in[0];
  const float* in_proj_w= (const float*)d_in[1];
  const float* in_proj_b= (const float*)d_in[2];
  const float* out_w    = (const float*)d_in[3];
  const float* out_b    = (const float*)d_in[4];
  const float* ln1_g    = (const float*)d_in[5];
  const float* ln1_b    = (const float*)d_in[6];
  const float* w1       = (const float*)d_in[7];
  const float* b1       = (const float*)d_in[8];
  const float* w2       = (const float*)d_in[9];
  const float* b2       = (const float*)d_in[10];
  const float* ln2_g    = (const float*)d_in[11];
  const float* ln2_b    = (const float*)d_in[12];

  // workspace carve-up (256B aligned)
  char* ws = (char*)d_ws;
  size_t off = 0;
  auto carve = [&](size_t bytes) {
    char* p = ws + off;
    off = (off + bytes + 255) & ~(size_t)255;
    return p;
  };
  bf16_t* Wqkv = (bf16_t*)carve((size_t)D3_  * DD_  * 2);
  bf16_t* Wout = (bf16_t*)carve((size_t)DD_  * DD_  * 2);
  bf16_t* W1   = (bf16_t*)carve((size_t)DFF_ * DD_  * 2);
  bf16_t* W2   = (bf16_t*)carve((size_t)DD_  * DFF_ * 2);
  bf16_t* hbuf = (bf16_t*)carve((size_t)NR_  * DD_  * 2);   // h -> o -> h2
  bf16_t* big  = (bf16_t*)carve((size_t)NR_  * DFF_ * 2);   // qkv -> ff
  float*  x1   = (float*) carve((size_t)NR_  * DD_  * 4);
  bf16_t* vT   = (bf16_t*)carve((size_t)BB_ * HH_ * HD_ * TT_ * 2);
  (void)ws_size; (void)in_sizes; (void)n_in; (void)out_size;

  // 1) weights -> bf16 (then resident in L2 for all GEMMs)
  auto cvt = [&](const float* src, bf16_t* dst, int n) {
    cvt_f32_bf16<<<(n + 255) / 256, 256, 0, stream>>>(src, dst, n);
  };
  cvt(in_proj_w, Wqkv, D3_ * DD_);
  cvt(out_w,     Wout, DD_ * DD_);
  cvt(w1,        W1,   DFF_ * DD_);
  cvt(w2,        W2,   DD_ * DFF_);

  // 2) h = LN1(x) (bf16)
  ln_bf16<<<NR_, 256, 0, stream>>>(x, ln1_g, ln1_b, hbuf);

  // 3) qkv = h @ Wqkv^T + b (bf16)
  gemm_wmma<0><<<dim3(D3_ / 128, NR_ / 128), 256, 0, stream>>>(
      hbuf, Wqkv, in_proj_b, nullptr, nullptr, big, NR_, D3_, DD_);

  // 4a) per-head V transpose for contiguous P.V fragments
  transpose_v<<<BB_ * HH_ * (TT_ / 64), 256, 0, stream>>>(big, vT);

  // 4b) flash look-ahead attention -> o (bf16, reuses hbuf)
  attn_wmma<<<BB_ * HH_ * (TT_ / 32) / 8, 256, 0, stream>>>(big, vT, hbuf);

  // 5) x1 = x + o @ Wout^T + b (f32)
  gemm_wmma<1><<<dim3(DD_ / 128, NR_ / 128), 256, 0, stream>>>(
      hbuf, Wout, out_b, x, x1, nullptr, NR_, DD_, DD_);

  // 6) h2 = LN2(x1) (bf16, reuses hbuf)
  ln_bf16<<<NR_, 256, 0, stream>>>(x1, ln2_g, ln2_b, hbuf);

  // 7) ff = GELU(h2 @ W1^T + b1) (bf16, reuses big)
  gemm_wmma<2><<<dim3(DFF_ / 128, NR_ / 128), 256, 0, stream>>>(
      hbuf, W1, b1, nullptr, nullptr, big, NR_, DFF_, DD_);

  // 8) out = x1 + ff @ W2^T + b2 (f32)
  gemm_wmma<1><<<dim3(DD_ / 128, NR_ / 128), 256, 0, stream>>>(
      big, W2, b2, x1, (float*)d_out, nullptr, NR_, DD_, DFF_);
}
